// NGramLanguageModel_7009386627386
// MI455X (gfx1250) — matrix operand order repaired and checked by reading, not compiled, not run on hardware
//
#include <hip/hip_runtime.h>

// ---------------- model constants ----------------
constexpr int CB = 2;
constexpr int CT = 2048;
constexpr int CD = 1024;
constexpr int CH = 16;      // heads
constexpr int CHD = 64;     // head dim
constexpr int CBT = CB * CT;        // 4096 rows
constexpr int CD4 = 4 * CD;         // 4096
constexpr int CV = 32000;

typedef __attribute__((ext_vector_type(16))) _Float16 v16h;
typedef __attribute__((ext_vector_type(8)))  float    v8f;
typedef __attribute__((ext_vector_type(4)))  unsigned int tdm_u32x4;
typedef __attribute__((ext_vector_type(8)))  int          tdm_i32x8;
typedef __attribute__((ext_vector_type(4)))  int          tdm_i32x4;

// ---------------- helpers ----------------
__device__ __forceinline__ unsigned short f2h(float f) {
  union { _Float16 h; unsigned short u; } c;
  c.h = (_Float16)f;
  return c.u;
}

// Tensor Data Mover: load a tile_d1 x tile_d0 (rows x contiguous-K) tile of a
// 2D f16 tensor [dim1][dim0] (row stride = stride0 elements) into LDS.
// D# layout per CDNA5 ISA ch.8 (group0 128b, group1 256b; groups 2/3 zero).
__device__ __forceinline__ void tdm_load_2d_f16(void* lds_ptr,
                                                const void* gptr,
                                                unsigned dim0, unsigned dim1,
                                                unsigned tile_d0, unsigned tile_d1,
                                                unsigned long long stride0) {
  unsigned lds_off = (unsigned)(unsigned long long)(size_t)lds_ptr;  // low 32 bits = LDS byte offset
  unsigned long long ga = (unsigned long long)(size_t)gptr;
  tdm_u32x4 g0;
  g0[0] = 1u;                                            // count=1 (valid), no gather
  g0[1] = lds_off;                                       // lds_addr
  g0[2] = (unsigned)(ga & 0xffffffffu);                  // global_addr[31:0]
  g0[3] = (unsigned)((ga >> 32) & 0x01ffffffu) | 0x80000000u;  // addr[56:32] | type=2
  tdm_i32x8 g1;
  g1[0] = 0x10000;                                       // data_size=1 (2 bytes), mask=0
  g1[1] = (int)((dim0 & 0xffffu) << 16);                 // tensor_dim0[15:0] @ bits 63:48
  g1[2] = (int)(((dim0 >> 16) & 0xffffu) | ((dim1 & 0xffffu) << 16));
  g1[3] = (int)(((dim1 >> 16) & 0xffffu) | ((tile_d0 & 0xffffu) << 16));
  g1[4] = (int)(tile_d1 & 0xffffu);                      // tile_dim1 (tile_dim2=0)
  g1[5] = (int)(stride0 & 0xffffffffull);                // tensor_dim0_stride lo32
  g1[6] = (int)((stride0 >> 32) & 0xffffull);            // stride hi16 (dim1_stride=0)
  g1[7] = 0;
  tdm_i32x4 z4 = {0, 0, 0, 0};
#if defined(__clang_major__) && (__clang_major__ >= 23)
  tdm_i32x8 z8 = {0, 0, 0, 0, 0, 0, 0, 0};
  __builtin_amdgcn_tensor_load_to_lds(g0, g1, z4, z4, z8, 0);
#else
  __builtin_amdgcn_tensor_load_to_lds(g0, g1, z4, z4, 0);
#endif
}

// A-matrix fragment, 16x32 f16, source stored row-major [m][k] (stride in halves).
// ISA layout: lanes 0-15 hold M=lane, K=0..7 (v0..3) and 16..23 (v4..7);
// lanes 16-31 hold K=8..15 and 24..31.
__device__ __forceinline__ v16h frag_a(const unsigned short* base, int stride, int lane) {
  union { v16h v; unsigned int w[8]; } u;
  const unsigned short* p = base + (size_t)(lane & 15) * stride + ((lane >> 4) << 3);
#pragma unroll
  for (int i = 0; i < 8; ++i) {
    int kp = (i < 4) ? (2 * i) : (8 + 2 * i);
    u.w[i] = *(const unsigned int*)(p + kp);
  }
  return u.v;
}

// B-matrix fragment, 32x16 f16, source stored as [n][k] row-major (stride in halves).
// ISA layout: lanes 0-15 hold N=lane, K=0..7 and 16..23; lanes 16-31 K=8..15 and 24..31.
__device__ __forceinline__ v16h frag_b(const unsigned short* base, int stride, int lane) {
  union { v16h v; uint4 q[2]; } u;
  const unsigned short* p = base + (size_t)(lane & 15) * stride + ((lane >> 4) << 3);
  u.q[0] = *(const uint4*)(p);
  u.q[1] = *(const uint4*)(p + 16);
  return u.v;
}

#define WMMA_F16(a, b, c) \
  __builtin_amdgcn_wmma_f32_16x16x32_f16(false, (a), false, (b), (short)0, (c), false, false)

// ---------------- embedding ----------------
__global__ void __launch_bounds__(256) embed_kernel(const int* __restrict__ idx,
                                                    const float* __restrict__ tok,
                                                    const float* __restrict__ pos,
                                                    float* __restrict__ x) {
  int i = blockIdx.x * 256 + threadIdx.x;
  if (i >= CBT * CD) return;
  int d = i & (CD - 1);
  int bt = i >> 10;           // /CD
  int t = bt & (CT - 1);
  x[i] = tok[(size_t)idx[bt] * CD + d] + pos[(size_t)t * CD + d];
}

// ---------------- weight convert+transpose: f32 [K][N] -> f16 [N][K] ----------------
__global__ void __launch_bounds__(256) convw_kernel(const float* __restrict__ W,
                                                    unsigned short* __restrict__ Wt,
                                                    int K, int N) {
  __shared__ float tile[32][33];
  int k0 = blockIdx.y * 32, n0 = blockIdx.x * 32;
  int tx = threadIdx.x & 31, ty = threadIdx.x >> 5;   // 32 x 8
#pragma unroll
  for (int i = 0; i < 4; ++i)
    tile[ty + i * 8][tx] = W[(size_t)(k0 + ty + i * 8) * N + n0 + tx];
  __syncthreads();
#pragma unroll
  for (int i = 0; i < 4; ++i)
    Wt[(size_t)(n0 + ty + i * 8) * K + k0 + tx] = f2h(tile[tx][ty + i * 8]);
}

// ---------------- layernorm (f32 in -> f16 out) ----------------
__global__ void __launch_bounds__(256) ln_kernel(const float* __restrict__ x,
                                                 const float* __restrict__ g,
                                                 const float* __restrict__ b,
                                                 unsigned short* __restrict__ out,
                                                 int Dd) {
  __shared__ float red[256];
  int row = blockIdx.x, tid = threadIdx.x;
  const float* xr = x + (size_t)row * Dd;
  float s = 0.f;
  for (int i = tid; i < Dd; i += 256) s += xr[i];
  red[tid] = s; __syncthreads();
  for (int st = 128; st > 0; st >>= 1) { if (tid < st) red[tid] += red[tid + st]; __syncthreads(); }
  float mu = red[0] / Dd;
  __syncthreads();
  float v = 0.f;
  for (int i = tid; i < Dd; i += 256) { float d = xr[i] - mu; v += d * d; }
  red[tid] = v; __syncthreads();
  for (int st = 128; st > 0; st >>= 1) { if (tid < st) red[tid] += red[tid + st]; __syncthreads(); }
  float rstd = rsqrtf(red[0] / Dd + 1e-5f);
  for (int i = tid; i < Dd; i += 256)
    out[(size_t)row * Dd + i] = f2h((xr[i] - mu) * rstd * g[i] + b[i]);
}

// ---------------- generic WMMA GEMM with TDM double-buffered staging ----------------
// C[M,N] = A_f16[M,K] @ Wt_f16[N][K]^T (+bias), epilogue by mode:
//   0: f32 store  1: f32 residual add  2: f16 relu store  3: f16 store
//   4: f16 store transposed (out[n*M + m])
// Block = 128 threads (4 waves), block tile 128x128, wave tile 64x64, K step 32.
__global__ void __launch_bounds__(128) gemm_f16(const unsigned short* __restrict__ A,
                                                const unsigned short* __restrict__ Wt,
                                                const float* __restrict__ bias,
                                                float* __restrict__ outF,
                                                unsigned short* __restrict__ outH,
                                                int M, int N, int K, int mode) {
  __shared__ unsigned short lA[2][128 * 32];   // [m][k] double-buffered
  __shared__ unsigned short lB[2][128 * 32];   // [n][k] double-buffered
  const int tid = threadIdx.x, lane = tid & 31, wid = tid >> 5;   // 4 waves
  const int m0 = blockIdx.y * 128, n0 = blockIdx.x * 128;
  const int wm = (wid >> 1) * 64, wn = (wid & 1) * 64;

  v8f acc[4][4];
#pragma unroll
  for (int mi = 0; mi < 4; ++mi)
#pragma unroll
    for (int ni = 0; ni < 4; ++ni) acc[mi][ni] = (v8f){};

  const unsigned short* Abase = A + (size_t)m0 * K;
  const unsigned short* Bbase = Wt + (size_t)n0 * K;

  if (wid == 0) {
    tdm_load_2d_f16(&lA[0][0], Abase, (unsigned)K, (unsigned)M, 32u, 128u, (unsigned long long)K);
    tdm_load_2d_f16(&lB[0][0], Bbase, (unsigned)K, (unsigned)N, 32u, 128u, (unsigned long long)K);
  }

  const int nsteps = K >> 5;
  for (int s = 0; s < nsteps; ++s) {
    const int cur = s & 1;
    if (wid == 0) __builtin_amdgcn_s_wait_tensorcnt(0);
    __syncthreads();                       // buffer `cur` ready for everyone
    if (wid == 0 && (s + 1) < nsteps) {    // prefetch next tile into the other buffer
      const int k1 = (s + 1) << 5;
      tdm_load_2d_f16(&lA[cur ^ 1][0], Abase + k1, (unsigned)K, (unsigned)M, 32u, 128u,
                      (unsigned long long)K);
      tdm_load_2d_f16(&lB[cur ^ 1][0], Bbase + k1, (unsigned)K, (unsigned)N, 32u, 128u,
                      (unsigned long long)K);
    }

    v16h af[4], bf[4];
#pragma unroll
    for (int mi = 0; mi < 4; ++mi) af[mi] = frag_a(&lA[cur][0] + (wm + mi * 16) * 32, 32, lane);
#pragma unroll
    for (int ni = 0; ni < 4; ++ni) bf[ni] = frag_b(&lB[cur][0] + (wn + ni * 16) * 32, 32, lane);
#pragma unroll
    for (int mi = 0; mi < 4; ++mi)
#pragma unroll
      for (int ni = 0; ni < 4; ++ni)
        acc[mi][ni] = WMMA_F16(af[mi], bf[ni], acc[mi][ni]);
  }

  // epilogue
  const int col16 = lane & 15, hi8 = (lane >> 4) * 8;
#pragma unroll
  for (int mi = 0; mi < 4; ++mi)
#pragma unroll
    for (int ni = 0; ni < 4; ++ni) {
      int n = n0 + wn + ni * 16 + col16;
      float bv = bias ? bias[n] : 0.f;
#pragma unroll
      for (int r = 0; r < 8; ++r) {
        int m = m0 + wm + mi * 16 + hi8 + r;
        float v = acc[mi][ni][r] + bv;
        if (mode == 0)      outF[(size_t)m * N + n] = v;
        else if (mode == 1) outF[(size_t)m * N + n] += v;
        else if (mode == 2) outH[(size_t)m * N + n] = f2h(fmaxf(v, 0.f));
        else if (mode == 3) outH[(size_t)m * N + n] = f2h(v);
        else                outH[(size_t)n * M + m] = f2h(v);   // mode 4: transposed
      }
    }
}

// ---------------- WMMA flash attention ----------------
// One wave per (b, h, 16-query tile); 32-key steps; online softmax; O -> f16.
// Vt is stored transposed: [d][bt] (d = h*64+hd), so PV B-fragments are contiguous.
__global__ void __launch_bounds__(128) attn_kernel(const unsigned short* __restrict__ Q,
                                                   const unsigned short* __restrict__ Kb,
                                                   const unsigned short* __restrict__ Vt,
                                                   unsigned short* __restrict__ O) {
  __shared__ unsigned short lP[4][16 * 32];   // per-wave P tile [m][k]
  const int lane = threadIdx.x & 31, wid = threadIdx.x >> 5;
  const int w = blockIdx.x * 4 + wid;
  const int QT = CT / 16;                      // 128 query tiles per (b,h)
  const int b = w / (CH * QT);
  const int rem = w % (CH * QT);
  const int h = rem / QT;
  const int qt = rem % QT;

  const unsigned short* qrow = Q + ((size_t)(b * CT + qt * 16)) * CD + h * CHD;
  const v16h aq0 = frag_a(qrow, CD, lane);
  const v16h aq1 = frag_a(qrow + 32, CD, lane);

  v8f o[4];
#pragma unroll
  for (int ni = 0; ni < 4; ++ni) o[ni] = (v8f){};
  float mrow[8], lrow[8];
#pragma unroll
  for (int r = 0; r < 8; ++r) { mrow[r] = -1e30f; lrow[r] = 0.f; }

  const int col = lane & 15, hi8 = (lane >> 4) * 8;
  const float scale = 0.03125f;               // D^-0.5 = 1/32
  const int kt2max = (qt * 16 + 15) >> 5;

  for (int kt2 = 0; kt2 <= kt2max; ++kt2) {
    const int key0 = kt2 * 32;
    // S = Q @ K^T for 32 keys (two 16x16 tiles)
    v8f S[2];
#pragma unroll
    for (int s = 0; s < 2; ++s) {
      const unsigned short* krow = Kb + ((size_t)(b * CT + key0 + s * 16)) * CD + h * CHD;
      v16h bk0 = frag_b(krow, CD, lane);       // K rows are exactly [n][k]
      v16h bk1 = frag_b(krow + 32, CD, lane);
      S[s] = (v8f){};
      S[s] = WMMA_F16(aq0, bk0, S[s]);
      S[s] = WMMA_F16(aq1, bk1, S[s]);
    }
    // scale, causal mask, online softmax update
#pragma unroll
    for (int r = 0; r < 8; ++r) {
      int qi = qt * 16 + hi8 + r;
      float s0 = S[0][r] * scale;
      float s1 = S[1][r] * scale;
      if (key0 + col > qi)      s0 = -1e30f;
      if (key0 + 16 + col > qi) s1 = -1e30f;
      float rmax = fmaxf(s0, s1);
      rmax = fmaxf(rmax, __shfl_xor(rmax, 1, 32));
      rmax = fmaxf(rmax, __shfl_xor(rmax, 2, 32));
      rmax = fmaxf(rmax, __shfl_xor(rmax, 4, 32));
      rmax = fmaxf(rmax, __shfl_xor(rmax, 8, 32));
      float mnew = fmaxf(mrow[r], rmax);
      float alpha = __expf(mrow[r] - mnew);
      float p0 = __expf(s0 - mnew);
      float p1 = __expf(s1 - mnew);
      float rsum = p0 + p1;
      rsum += __shfl_xor(rsum, 1, 32);
      rsum += __shfl_xor(rsum, 2, 32);
      rsum += __shfl_xor(rsum, 4, 32);
      rsum += __shfl_xor(rsum, 8, 32);
      lrow[r] = lrow[r] * alpha + rsum;
      mrow[r] = mnew;
#pragma unroll
      for (int ni = 0; ni < 4; ++ni) o[ni][r] *= alpha;
      lP[wid][(hi8 + r) * 32 + col] = f2h(p0);
      lP[wid][(hi8 + r) * 32 + 16 + col] = f2h(p1);
    }
    asm volatile("s_wait_dscnt 0x0" ::: "memory");   // P tile visible to own wave
    v16h ap = frag_a(&lP[wid][0], 32, lane);
    // V fragments: Vt row = d, contiguous keys along the row
    const unsigned short* vbase = Vt + (size_t)(h * CHD) * CBT + (b * CT + key0);
#pragma unroll
    for (int ni = 0; ni < 4; ++ni) {
      v16h bv = frag_b(vbase + (size_t)(ni * 16) * CBT, CBT, lane);
      o[ni] = WMMA_F16(ap, bv, o[ni]);
    }
  }

  // normalize and store f16
#pragma unroll
  for (int r = 0; r < 8; ++r) {
    float inv = 1.f / lrow[r];
    size_t rowoff = ((size_t)(b * CT + qt * 16 + hi8 + r)) * CD + h * CHD;
#pragma unroll
    for (int ni = 0; ni < 4; ++ni)
      O[rowoff + ni * 16 + col] = f2h(o[ni][r] * inv);
  }
}

// ---------------- loss ----------------
__global__ void loss_zero(float* p) { if (threadIdx.x == 0 && blockIdx.x == 0) *p = 0.f; }

__global__ void __launch_bounds__(256) nll_kernel(const float* __restrict__ logits,
                                                  const int* __restrict__ tgt,
                                                  float* __restrict__ loss) {
  __shared__ float red[256];
  int row = blockIdx.x, tid = threadIdx.x;
  const float* lr = logits + (size_t)row * CV;
  float mx = -1e30f;
  for (int i = tid; i < CV; i += 256) mx = fmaxf(mx, lr[i]);
  red[tid] = mx; __syncthreads();
  for (int st = 128; st > 0; st >>= 1) { if (tid < st) red[tid] = fmaxf(red[tid], red[tid + st]); __syncthreads(); }
  mx = red[0];
  __syncthreads();
  float s = 0.f;
  for (int i = tid; i < CV; i += 256) s += __expf(lr[i] - mx);
  red[tid] = s; __syncthreads();
  for (int st = 128; st > 0; st >>= 1) { if (tid < st) red[tid] += red[tid + st]; __syncthreads(); }
  if (tid == 0) {
    float lse = mx + logf(red[0]);
    float v = lse - lr[tgt[row]];
    atomicAdd(loss, v / (float)CBT);
  }
}

// ---------------- host driver ----------------
static void launch_gemm(hipStream_t stream, const unsigned short* A, const unsigned short* Wt,
                        const float* bias, float* oF, unsigned short* oH,
                        int M, int N, int K, int mode) {
  dim3 grid(N / 128, M / 128);
  gemm_f16<<<grid, 128, 0, stream>>>(A, Wt, bias, oF, oH, M, N, K, mode);
}

static void launch_convw(hipStream_t stream, const float* W, unsigned short* Wt, int K, int N) {
  dim3 grid(N / 32, K / 32);
  convw_kernel<<<grid, 256, 0, stream>>>(W, Wt, K, N);
}

extern "C" void kernel_launch(void* const* d_in, const int* in_sizes, int n_in,
                              void* d_out, int out_size, void* d_ws, size_t ws_size,
                              hipStream_t stream) {
  (void)in_sizes; (void)n_in; (void)out_size; (void)ws_size;
  const int*   idx     = (const int*)d_in[0];
  const int*   targets = (const int*)d_in[1];
  const float* tok_emb = (const float*)d_in[2];
  const float* pos_emb = (const float*)d_in[3];
  const float* Wq      = (const float*)d_in[4];
  const float* Wk      = (const float*)d_in[5];
  const float* Wv      = (const float*)d_in[6];
  const float* Wo      = (const float*)d_in[7];
  const float* bo      = (const float*)d_in[8];
  const float* ln1_g   = (const float*)d_in[9];
  const float* ln1_b   = (const float*)d_in[10];
  const float* ln2_g   = (const float*)d_in[11];
  const float* ln2_b   = (const float*)d_in[12];
  const float* w1      = (const float*)d_in[13];
  const float* b1      = (const float*)d_in[14];
  const float* w2      = (const float*)d_in[15];
  const float* b2      = (const float*)d_in[16];
  const float* lnf_g   = (const float*)d_in[17];
  const float* lnf_b   = (const float*)d_in[18];
  const float* lm_w    = (const float*)d_in[19];
  const float* lm_b    = (const float*)d_in[20];

  // workspace layout (~146 MB total)
  char* ws = (char*)d_ws;
  float*          xf = (float*)ws;                                   // 16 MB  x (f32) [4096,1024]
  unsigned short* a0 = (unsigned short*)(ws + ((size_t)16 << 20));   //  8 MB  f16 [4096,1024]
  unsigned short* a1 = (unsigned short*)(ws + ((size_t)24 << 20));   // 32 MB  f16 [4096,4096]
  unsigned short* qb = (unsigned short*)(ws + ((size_t)56 << 20));   //  8 MB
  unsigned short* kb = (unsigned short*)(ws + ((size_t)64 << 20));   //  8 MB
  unsigned short* vb = (unsigned short*)(ws + ((size_t)72 << 20));   //  8 MB  (transposed [d][bt])
  unsigned short* wt = (unsigned short*)(ws + ((size_t)80 << 20));   // 63 MB  f16 [N][K] scratch

  embed_kernel<<<(CBT * CD) / 256, 256, 0, stream>>>(idx, tok_emb, pos_emb, xf);

  for (int l = 0; l < 8; ++l) {
    const float* wq = Wq + (size_t)l * CD * CD;
    const float* wk = Wk + (size_t)l * CD * CD;
    const float* wv = Wv + (size_t)l * CD * CD;
    const float* wo = Wo + (size_t)l * CD * CD;
    const float* W1 = w1 + (size_t)l * CD * CD4;
    const float* W2 = w2 + (size_t)l * CD4 * CD;

    ln_kernel<<<CBT, 256, 0, stream>>>(xf, ln1_g + l * CD, ln1_b + l * CD, a0, CD);
    launch_convw(stream, wq, wt, CD, CD);
    launch_gemm(stream, a0, wt, nullptr, nullptr, qb, CBT, CD, CD, 3);
    launch_convw(stream, wk, wt, CD, CD);
    launch_gemm(stream, a0, wt, nullptr, nullptr, kb, CBT, CD, CD, 3);
    launch_convw(stream, wv, wt, CD, CD);
    launch_gemm(stream, a0, wt, nullptr, nullptr, vb, CBT, CD, CD, 4);  // V transposed
    attn_kernel<<<(CB * CH * (CT / 16)) / 4, 128, 0, stream>>>(qb, kb, vb, a0);
    launch_convw(stream, wo, wt, CD, CD);
    launch_gemm(stream, a0, wt, bo + l * CD, xf, nullptr, CBT, CD, CD, 1);
    ln_kernel<<<CBT, 256, 0, stream>>>(xf, ln2_g + l * CD, ln2_b + l * CD, a0, CD);
    launch_convw(stream, W1, wt, CD, CD4);
    launch_gemm(stream, a0, wt, b1 + (size_t)l * CD4, nullptr, a1, CBT, CD4, CD, 2);
    launch_convw(stream, W2, wt, CD4, CD);
    launch_gemm(stream, a1, wt, b2 + l * CD, xf, nullptr, CBT, CD, CD4, 1);
  }

  ln_kernel<<<CBT, 256, 0, stream>>>(xf, lnf_g, lnf_b, a0, CD);

  float* logits = (float*)d_out;
  launch_convw(stream, lm_w, wt, CD, CV);
  launch_gemm(stream, a0, wt, lm_b, logits, nullptr, CBT, CV, CD, 0);

  float* lossp = logits + (size_t)CBT * CV;
  loss_zero<<<1, 1, 0, stream>>>(lossp);
  nll_kernel<<<CBT, 256, 0, stream>>>(logits, targets, lossp);
}
